// LSTM_26637387170353
// MI455X (gfx1250) — compile-verified, matrix operand order checked by reading, not run on hardware
//
#include <hip/hip_runtime.h>
#include <hip/hip_bf16.h>

// ---------------------------------------------------------------------------
// 2-layer LSTM (IN=512, H=1024, B=64, T=128) + FC(256) for MI455X / gfx1250.
// bf16 WMMA (v_wmma_f32_16x16x32_bf16) with fp32 accumulate; fp32 cell math.
// GEMM: two-phase K loop (no per-iteration pointer selects), 2 N-tiles/wave.
// ---------------------------------------------------------------------------

typedef __bf16 bf16_t;
typedef __attribute__((ext_vector_type(16))) __bf16 v16bf;
typedef __attribute__((ext_vector_type(8)))  __bf16 v8bf;
typedef __attribute__((ext_vector_type(8)))  float  v8f;

#define IN_DIM 512
#define HDIM   1024
#define GDIM   4096   // 4*H
#define OUTDIM 256
#define BDIM   64
#define TDIM   128
#define NT     2      // N-tiles per wave

__device__ __forceinline__ bf16_t f2bf(float f) {
  unsigned u = __builtin_bit_cast(unsigned, f);
  u += 0x7FFFu + ((u >> 16) & 1u);            // round to nearest even
  unsigned short s = (unsigned short)(u >> 16);
  return __builtin_bit_cast(bf16_t, s);
}

__device__ __forceinline__ v16bf cat8(v8bf lo, v8bf hi) {
  return __builtin_shufflevector(lo, hi, 0,1,2,3,4,5,6,7,8,9,10,11,12,13,14,15);
}

__global__ void cvt_f32_to_bf16(const float* __restrict__ src,
                                bf16_t* __restrict__ dst, int n) {
  int i = blockIdx.x * blockDim.x + threadIdx.x;
  if (i < n) dst[i] = f2bf(src[i]);
}

__global__ void zero_f32(float* __restrict__ p, int n) {
  int i = blockIdx.x * blockDim.x + threadIdx.x;
  if (i < n) p[i] = 0.0f;
}

__global__ void zero_bf16(bf16_t* __restrict__ p, int n) {
  int i = blockIdx.x * blockDim.x + threadIdx.x;
  if (i < n) p[i] = __builtin_bit_cast(bf16_t, (unsigned short)0);
}

// C[M,N] = concat_K(A0,A1) * concat_K(W0,W1)^T + bias0[n] + bias1[n]
// Phase 1 runs K1 elements over A0/W0; phase 2 runs K2 elements over A1/W1.
// One wave computes a 16x(16*NT) slab: one A fragment, NT B fragments, NT wmma
// per 32-wide K step.  Per-lane fragment addressing follows the CDNA5 16-bit
// A/B layout: lanes 0-15 hold K-chunks {k..k+7, k+16..k+23}, lanes 16-31 the
// {+8, +24} chunks.  All loads are 16B-aligned global_load_b128.
__global__ void wmma_gemm_bias(const bf16_t* __restrict__ A0, int lda0,
                               const bf16_t* __restrict__ A1, int lda1,
                               const bf16_t* __restrict__ W0, int ldw0,
                               const bf16_t* __restrict__ W1, int ldw1,
                               const float* __restrict__ bias0,
                               const float* __restrict__ bias1,
                               float* __restrict__ C,
                               int M, int N, int K1, int K2) {
  const int wave    = (blockIdx.x * blockDim.x + threadIdx.x) >> 5;
  const int tilesM  = M >> 4;
  const int ngroups = (N >> 4) / NT;
  if (wave >= tilesM * ngroups) return;  // wave-uniform -> EXEC stays all-1s

  const int tileM  = wave % tilesM;
  const int groupN = wave / tilesM;
  const int lane   = threadIdx.x & 31;
  const int l15    = lane & 15;
  const int hi     = lane >> 4;          // 0 or 1
  const int koff   = hi * 8;

  const long arow = (long)(tileM * 16 + l15);
  const long wrow = (long)(groupN * (16 * NT) + l15);

  v8f acc0 = {};
  v8f acc1 = {};

  // ---- phase 1: K1 elements from A0 / W0 ----------------------------------
  {
    const bf16_t* pa  = A0 + arow * lda0 + koff;
    const bf16_t* pb0 = W0 + wrow * ldw0 + koff;
    const bf16_t* pb1 = W0 + (wrow + 16) * ldw0 + koff;
    for (int k = 0; k < K1; k += 32) {
      v16bf av = cat8(*(const v8bf*)(pa  + k), *(const v8bf*)(pa  + k + 16));
      v16bf b0 = cat8(*(const v8bf*)(pb0 + k), *(const v8bf*)(pb0 + k + 16));
      v16bf b1 = cat8(*(const v8bf*)(pb1 + k), *(const v8bf*)(pb1 + k + 16));
      __builtin_prefetch(pb0 + k + 2048, 0, 1);   // global_prefetch_b8, 4KB ahead
      acc0 = __builtin_amdgcn_wmma_f32_16x16x32_bf16(
          false, av, false, b0, (short)0, acc0, false, false);
      acc1 = __builtin_amdgcn_wmma_f32_16x16x32_bf16(
          false, av, false, b1, (short)0, acc1, false, false);
    }
  }

  // ---- phase 2: K2 elements from A1 / W1 ----------------------------------
  {
    const bf16_t* pa  = A1 + arow * lda1 + koff;
    const bf16_t* pb0 = W1 + wrow * ldw1 + koff;
    const bf16_t* pb1 = W1 + (wrow + 16) * ldw1 + koff;
    for (int k = 0; k < K2; k += 32) {
      v16bf av = cat8(*(const v8bf*)(pa  + k), *(const v8bf*)(pa  + k + 16));
      v16bf b0 = cat8(*(const v8bf*)(pb0 + k), *(const v8bf*)(pb0 + k + 16));
      v16bf b1 = cat8(*(const v8bf*)(pb1 + k), *(const v8bf*)(pb1 + k + 16));
      __builtin_prefetch(pb0 + k + 2048, 0, 1);
      acc0 = __builtin_amdgcn_wmma_f32_16x16x32_bf16(
          false, av, false, b0, (short)0, acc0, false, false);
      acc1 = __builtin_amdgcn_wmma_f32_16x16x32_bf16(
          false, av, false, b1, (short)0, acc1, false, false);
    }
  }

  // ---- epilogue: bias add + fp32 store -------------------------------------
  const int n0 = groupN * (16 * NT) + l15;
  const int n1 = n0 + 16;
  float badd0 = 0.0f, badd1 = 0.0f;
  if (bias0) { badd0 += bias0[n0]; badd1 += bias0[n1]; }
  if (bias1) { badd0 += bias1[n0]; badd1 += bias1[n1]; }
#pragma unroll
  for (int r = 0; r < 8; ++r) {
    const int m = tileM * 16 + hi * 8 + r;   // C/D layout: VGPR r -> M=r (+8 hi half)
    C[(long)m * N + n0] = acc0[r] + badd0;
    C[(long)m * N + n1] = acc1[r] + badd1;
  }
}

// Elementwise LSTM cell: gates (B x 4H fp32, order i|f|g|o) -> c (fp32), h (bf16)
__global__ void lstm_cell(const float* __restrict__ gates,
                          float* __restrict__ c,
                          bf16_t* __restrict__ hprev,
                          bf16_t* __restrict__ hseq,   // nullable
                          long hseq_stride) {
  int i = blockIdx.x * blockDim.x + threadIdx.x;   // 0 .. B*H-1
  if (i >= BDIM * HDIM) return;
  int b = i >> 10;          // / HDIM
  int n = i & (HDIM - 1);
  const float* g = gates + (long)b * GDIM;
  float ig = g[n];
  float fg = g[n + HDIM];
  float gg = g[n + 2 * HDIM];
  float og = g[n + 3 * HDIM];
  float si = 1.0f / (1.0f + __expf(-ig));
  float sf = 1.0f / (1.0f + __expf(-fg));
  float so = 1.0f / (1.0f + __expf(-og));
  float tg = tanhf(gg);
  float cn = sf * c[i] + si * tg;
  c[i] = cn;
  float h = so * tanhf(cn);
  bf16_t hb = f2bf(h);
  hprev[i] = hb;
  if (hseq) hseq[(long)b * hseq_stride + n] = hb;
}

// ---------------------------------------------------------------------------

extern "C" void kernel_launch(void* const* d_in, const int* in_sizes, int n_in,
                              void* d_out, int out_size, void* d_ws, size_t ws_size,
                              hipStream_t stream) {
  const float* x    = (const float*)d_in[0];
  const float* Wxh0 = (const float*)d_in[1];
  const float* bxh0 = (const float*)d_in[2];
  const float* Whh0 = (const float*)d_in[3];
  const float* bhh0 = (const float*)d_in[4];
  const float* Wxh1 = (const float*)d_in[5];
  const float* bxh1 = (const float*)d_in[6];
  const float* Whh1 = (const float*)d_in[7];
  const float* bhh1 = (const float*)d_in[8];
  const float* Wfc  = (const float*)d_in[9];
  const float* bfc  = (const float*)d_in[10];
  float* out = (float*)d_out;

  char* ws = (char*)d_ws;
  size_t off = 0;
  auto alloc = [&](size_t bytes) -> void* {
    off = (off + 255) & ~(size_t)255;
    void* p = ws + off;
    off += bytes;
    return p;
  };

  bf16_t* xb     = (bf16_t*)alloc((size_t)BDIM * TDIM * IN_DIM * 2);
  bf16_t* Wxh0b  = (bf16_t*)alloc((size_t)GDIM * IN_DIM * 2);
  bf16_t* Whh0b  = (bf16_t*)alloc((size_t)GDIM * HDIM * 2);
  bf16_t* Wxh1b  = (bf16_t*)alloc((size_t)GDIM * HDIM * 2);
  bf16_t* Whh1b  = (bf16_t*)alloc((size_t)GDIM * HDIM * 2);
  bf16_t* Wfcb   = (bf16_t*)alloc((size_t)OUTDIM * HDIM * 2);
  bf16_t* h1b    = (bf16_t*)alloc((size_t)BDIM * TDIM * HDIM * 2); // layer0 out seq
  float*  gates  = (float*)alloc((size_t)BDIM * GDIM * 4);
  float*  cbuf   = (float*)alloc((size_t)BDIM * HDIM * 4);
  bf16_t* hprev  = (bf16_t*)alloc((size_t)BDIM * HDIM * 2);

  const int THREADS = 256;
  auto cdiv = [](int a, int b) { return (a + b - 1) / b; };

  // ---- fp32 -> bf16 conversions --------------------------------------------
  {
    int n;
    n = BDIM * TDIM * IN_DIM;
    cvt_f32_to_bf16<<<cdiv(n, THREADS), THREADS, 0, stream>>>(x, xb, n);
    n = GDIM * IN_DIM;
    cvt_f32_to_bf16<<<cdiv(n, THREADS), THREADS, 0, stream>>>(Wxh0, Wxh0b, n);
    n = GDIM * HDIM;
    cvt_f32_to_bf16<<<cdiv(n, THREADS), THREADS, 0, stream>>>(Whh0, Whh0b, n);
    cvt_f32_to_bf16<<<cdiv(n, THREADS), THREADS, 0, stream>>>(Wxh1, Wxh1b, n);
    cvt_f32_to_bf16<<<cdiv(n, THREADS), THREADS, 0, stream>>>(Whh1, Whh1b, n);
    n = OUTDIM * HDIM;
    cvt_f32_to_bf16<<<cdiv(n, THREADS), THREADS, 0, stream>>>(Wfc, Wfcb, n);
  }

  const int bh = BDIM * HDIM;
  const int cellBlocks = cdiv(bh, THREADS);

  // step GEMM: tilesM=4, ngroups=4096/16/NT=128 -> 512 waves -> 64 blocks
  const int stepWaves  = (BDIM / 16) * ((GDIM / 16) / NT);
  const int stepBlocks = (stepWaves * 32) / THREADS;

  // ---- layer 0 -------------------------------------------------------------
  zero_f32<<<cellBlocks, THREADS, 0, stream>>>(cbuf, bh);
  zero_bf16<<<cellBlocks, THREADS, 0, stream>>>(hprev, bh);
  for (int t = 0; t < TDIM; ++t) {
    wmma_gemm_bias<<<stepBlocks, THREADS, 0, stream>>>(
        xb + (size_t)t * IN_DIM, TDIM * IN_DIM,   // A0 = x[:, t, :]
        hprev, HDIM,                               // A1 = h_prev
        Wxh0b, IN_DIM,                             // W0
        Whh0b, HDIM,                               // W1
        bxh0, bhh0, gates,
        BDIM, GDIM, IN_DIM, HDIM);
    lstm_cell<<<cellBlocks, THREADS, 0, stream>>>(
        gates, cbuf, hprev,
        h1b + (size_t)t * HDIM, (long)TDIM * HDIM);
  }

  // ---- layer 1 -------------------------------------------------------------
  zero_f32<<<cellBlocks, THREADS, 0, stream>>>(cbuf, bh);
  zero_bf16<<<cellBlocks, THREADS, 0, stream>>>(hprev, bh);
  for (int t = 0; t < TDIM; ++t) {
    wmma_gemm_bias<<<stepBlocks, THREADS, 0, stream>>>(
        h1b + (size_t)t * HDIM, TDIM * HDIM,       // A0 = h1[:, t, :]
        hprev, HDIM,                               // A1 = h_prev
        Wxh1b, HDIM,                               // W0
        Whh1b, HDIM,                               // W1
        bxh1, bhh1, gates,
        BDIM, GDIM, HDIM, HDIM);
    lstm_cell<<<cellBlocks, THREADS, 0, stream>>>(
        gates, cbuf, hprev, nullptr, 0);
  }

  // ---- FC: out = h2_last @ Wfc^T + bfc ------------------------------------
  // tilesM=4, ngroups=256/16/NT=8 -> 32 waves -> 4 blocks
  {
    const int fcWaves  = (BDIM / 16) * ((OUTDIM / 16) / NT);
    const int fcBlocks = (fcWaves * 32) / THREADS;
    wmma_gemm_bias<<<fcBlocks, THREADS, 0, stream>>>(
        hprev, HDIM, hprev, HDIM,
        Wfcb, HDIM, Wfcb, HDIM,
        bfc, nullptr, out,
        BDIM, OUTDIM, HDIM, 0);
  }
}
